// MultiHeadGraphAttentionLayer_20246475833395
// MI455X (gfx1250) — matrix operand order repaired and checked by reading, not compiled, not run on hardware
//
#include <hip/hip_runtime.h>

// GAT layer, MI455X (gfx1250, wave32). Flash-style single pass over the
// 134MB adjacency (the bandwidth-dominant stream), bf16 WMMA for both the
// projection and the attn@h GEMM, fp32 accumulation.

#define B_     2
#define N_     4096
#define INDIM  512
#define H_     4
#define HD     64
#define LEAKY  0.01f

typedef __attribute__((ext_vector_type(16))) __bf16 v16bf;
typedef __attribute__((ext_vector_type(8)))  __bf16 v8bf;
typedef __attribute__((ext_vector_type(8)))  float  v8f;

// max/sum across the lane-halves that share a matrix row (lane l <-> l^16)
static __device__ __forceinline__ float swz_xor16(float x) {
  // ds_swizzle group-of-32: xor=0x10, or=0, and=0x1f -> offset 0x401f (SWAPX16)
  return __int_as_float(__builtin_amdgcn_ds_swizzle(__float_as_int(x), 0x401f));
}
// broadcast per-row value (held in lane == row) to C-fragment layout lanes
static __device__ __forceinline__ float bcast_row(int srclane, float x) {
  return __int_as_float(__builtin_amdgcn_ds_bpermute(srclane << 2, __float_as_int(x)));
}

// ---- kernel 0: W[H][K][D] (f32) -> Wt[H][D][K] (bf16), contiguous in K ----
__global__ void k_wt(const float* __restrict__ W, __bf16* __restrict__ Wt) {
  int idx = blockIdx.x * blockDim.x + threadIdx.x;   // H*HD*INDIM
  int k = idx & (INDIM - 1);
  int d = (idx >> 9) & (HD - 1);
  int h = idx >> 15;
  Wt[idx] = (__bf16)W[((size_t)(h * INDIM + k)) * HD + d];
}

// ---- kernel 1: hT = W^T * X^T per (b,h), 64x16 output tile per wave -------
// A-frag = Wt rows (M=d), B-frag = X rows (N=i, K contiguous in memory).
__global__ void __launch_bounds__(32) k_proj(const float* __restrict__ X,
                                             const __bf16* __restrict__ Wt,
                                             __bf16* __restrict__ hT) {
  const int wave = blockIdx.x;              // B*H*(N/16)
  const int lane = threadIdx.x & 31;
  const int row16 = lane & 15, half = lane >> 4;
  const int it = wave & (N_ / 16 - 1);
  const int bh = wave >> 8;                 // N/16 == 256
  const int b = bh >> 2, h = bh & (H_ - 1);
  const int i0 = it * 16;

  v8f acc[4];
#pragma unroll
  for (int t = 0; t < 4; ++t) acc[t] = (v8f)0.f;

  const float* xrow = X + ((size_t)(b * N_ + i0 + row16)) * INDIM;

  for (int k0 = 0; k0 < INDIM; k0 += 32) {
    // B fragment (K x N = k x i): lane holds i=i0+row16, K = 16*half + e
    const float4* xp = (const float4*)(xrow + k0 + 16 * half);
    float4 x0 = xp[0], x1 = xp[1], x2 = xp[2], x3 = xp[3];
    float xv[16] = {x0.x, x0.y, x0.z, x0.w, x1.x, x1.y, x1.z, x1.w,
                    x2.x, x2.y, x2.z, x2.w, x3.x, x3.y, x3.z, x3.w};
    v16bf bfrag;
#pragma unroll
    for (int e = 0; e < 16; ++e) bfrag[e] = (__bf16)xv[e];

#pragma unroll
    for (int t = 0; t < 4; ++t) {
      // A fragment (M x K = d x k): elem e -> K = 16*half + e
      int d = t * 16 + row16;
      const v8bf* wp =
          (const v8bf*)(Wt + ((size_t)(h * HD + d)) * INDIM + k0 + 16 * half);
      v8bf lo = wp[0];   // K offset 16*half + [0,8)
      v8bf hi = wp[1];   // K offset 16*half + 8 + [0,8)
      v16bf afrag;
#pragma unroll
      for (int e = 0; e < 8; ++e) { afrag[e] = lo[e]; afrag[e + 8] = hi[e]; }
      acc[t] = __builtin_amdgcn_wmma_f32_16x16x32_bf16(
          false, afrag, false, bfrag, (short)0, acc[t], false, false);
    }
  }
  // C layout: VGPR v, lane: M(=d_local) = v + 8*half, N(=i) = row16
#pragma unroll
  for (int t = 0; t < 4; ++t)
#pragma unroll
    for (int v = 0; v < 8; ++v) {
      int d = t * 16 + v + 8 * half;
      hT[((size_t)(bh * HD + d)) * N_ + i0 + row16] = (__bf16)acc[t][v];
    }
}

// ---- kernel 2: s_src/s_dst = <h_n, a_src/dst>, coalesced over n -----------
__global__ void k_scores(const __bf16* __restrict__ hT,
                         const float* __restrict__ a_src,
                         const float* __restrict__ a_dst,
                         float* __restrict__ s_src, float* __restrict__ s_dst) {
  int idx = blockIdx.x * blockDim.x + threadIdx.x;   // B*H*N
  int n = idx & (N_ - 1);
  int bh = idx >> 12;
  int h = bh & (H_ - 1);
  float ss = 0.f, sd = 0.f;
#pragma unroll 8
  for (int d = 0; d < HD; ++d) {
    float v = (float)hT[((size_t)(bh * HD + d)) * N_ + n];
    ss += v * a_src[h * HD + d];
    sd += v * a_dst[h * HD + d];
  }
  s_src[idx] = ss;
  s_dst[idx] = sd;
}

// ---- kernel 3: flash-style masked softmax + attn@h (bf16 WMMA) ------------
__global__ void __launch_bounds__(256) k_attn(const int* __restrict__ A,
                                              const __bf16* __restrict__ hT,
                                              const float* __restrict__ s_src,
                                              const float* __restrict__ s_dst,
                                              float* __restrict__ out) {
  const int lane = threadIdx.x & 31;
  const int warp = threadIdx.x >> 5;
  const int g = blockIdx.x * 8 + warp;      // B*H*(N/16) waves total
  const int it = g & (N_ / 16 - 1);
  const int bh = g >> 8;
  const int b = bh >> 2, h = bh & (H_ - 1);
  const int i0 = it * 16;
  const int row = lane & 15, half = lane >> 4;

  const int* arow = A + ((size_t)b * N_ + i0 + row) * N_;
  const float* sdrow = s_dst + (size_t)bh * N_;
  const __bf16* hbase = hT + (size_t)bh * HD * N_;
  const float ssr = s_src[(size_t)bh * N_ + i0 + row];

  float m = -1e30f, lsum = 0.f;
  v8f acc[4];
#pragma unroll
  for (int t = 0; t < 4; ++t) acc[t] = (v8f)0.f;

  const int klo = 16 * half;
  for (int j0 = 0; j0 < N_; j0 += 32) {
    if (j0 + 512 < N_) __builtin_prefetch(arow + j0 + 512, 0, 1);

    // lane's 16 K slots: K = 16*half + e (A-frag layout)
    int4 a0 = *(const int4*)(arow + j0 + klo);
    int4 a1 = *(const int4*)(arow + j0 + klo + 4);
    int4 a2 = *(const int4*)(arow + j0 + klo + 8);
    int4 a3 = *(const int4*)(arow + j0 + klo + 12);
    float4 s0 = *(const float4*)(sdrow + j0 + klo);
    float4 s1 = *(const float4*)(sdrow + j0 + klo + 4);
    float4 s2 = *(const float4*)(sdrow + j0 + klo + 8);
    float4 s3 = *(const float4*)(sdrow + j0 + klo + 12);
    int av[16] = {a0.x, a0.y, a0.z, a0.w, a1.x, a1.y, a1.z, a1.w,
                  a2.x, a2.y, a2.z, a2.w, a3.x, a3.y, a3.z, a3.w};
    float sv[16] = {s0.x, s0.y, s0.z, s0.w, s1.x, s1.y, s1.z, s1.w,
                    s2.x, s2.y, s2.z, s2.w, s3.x, s3.y, s3.z, s3.w};

    float ev[16];
    float cmax = -1e30f;
#pragma unroll
    for (int e = 0; e < 16; ++e) {
      float x = ssr + sv[e];
      x = x > 0.f ? x : LEAKY * x;          // leaky_relu before mask (ref order)
      x = av[e] > 0 ? x : -1e30f;           // mask non-edges
      ev[e] = x;
      cmax = fmaxf(cmax, x);
    }
    cmax = fmaxf(cmax, swz_xor16(cmax));    // combine lane-halves of the row
    float mnew = fmaxf(m, cmax);
    float scale = __expf(m - mnew);
    m = mnew;

    float psum = 0.f;
    v16bf pfrag;
#pragma unroll
    for (int e = 0; e < 16; ++e) {
      // guard: before any valid edge, mnew==-1e30 -> force p=0 (not exp(0))
      float p = (ev[e] > -1e29f) ? __expf(ev[e] - mnew) : 0.f;
      psum += p;
      pfrag[e] = (__bf16)p;
    }
    lsum = lsum * scale + psum;

    // rescale accumulator rows (C layout row = v + 8*half)
#pragma unroll
    for (int v = 0; v < 8; ++v) {
      float f = bcast_row(v + 8 * half, scale);
#pragma unroll
      for (int t = 0; t < 4; ++t) acc[t][v] *= f;
    }

    // B fragments from hT (K=j contiguous): lane d = 16t+row, j = j0+16*half+e
#pragma unroll
    for (int t = 0; t < 4; ++t) {
      v16bf bfrag =
          *(const v16bf*)(hbase + ((size_t)(t * 16 + row)) * N_ + j0 + 16 * half);
      acc[t] = __builtin_amdgcn_wmma_f32_16x16x32_bf16(
          false, pfrag, false, bfrag, (short)0, acc[t], false, false);
    }
  }

  float lrow = lsum + swz_xor16(lsum);      // full row sum
  float inv = 1.f / lrow;
  float osc[8];
#pragma unroll
  for (int v = 0; v < 8; ++v) osc[v] = bcast_row(v + 8 * half, inv);

  // out[b, i, h*64 + d]; C layout: M = v + 8*half, N(d) = 16t + row
#pragma unroll
  for (int t = 0; t < 4; ++t)
#pragma unroll
    for (int v = 0; v < 8; ++v) {
      int M = v + 8 * half;
      out[((size_t)(b * N_ + i0 + M)) * (H_ * HD) + h * HD + t * 16 + row] =
          acc[t][v] * osc[v];
    }
}

extern "C" void kernel_launch(void* const* d_in, const int* in_sizes, int n_in,
                              void* d_out, int out_size, void* d_ws, size_t ws_size,
                              hipStream_t stream) {
  const float* X     = (const float*)d_in[0];
  const int*   A     = (const int*)d_in[1];
  const float* W     = (const float*)d_in[2];
  const float* a_src = (const float*)d_in[3];
  const float* a_dst = (const float*)d_in[4];
  float* out = (float*)d_out;

  char* ws = (char*)d_ws;
  const size_t wt_bytes = (size_t)H_ * HD * INDIM * 2;      // 256 KB
  const size_t ht_bytes = (size_t)B_ * H_ * HD * N_ * 2;    // 4 MB
  const size_t sc_elems = (size_t)B_ * H_ * N_;
  __bf16* Wt   = (__bf16*)ws;
  __bf16* hT   = (__bf16*)(ws + wt_bytes);
  float*  ssrc = (float*)(ws + wt_bytes + ht_bytes);
  float*  sdst = ssrc + sc_elems;

  hipLaunchKernelGGL(k_wt, dim3((H_ * HD * INDIM) / 256), dim3(256), 0, stream,
                     W, Wt);
  hipLaunchKernelGGL(k_proj, dim3(B_ * H_ * (N_ / 16)), dim3(32), 0, stream,
                     X, Wt, hT);
  hipLaunchKernelGGL(k_scores, dim3((B_ * H_ * N_) / 256), dim3(256), 0, stream,
                     hT, a_src, a_dst, ssrc, sdst);
  hipLaunchKernelGGL(k_attn, dim3((B_ * H_ * (N_ / 16)) / 8), dim3(256), 0,
                     stream, A, hT, ssrc, sdst, out);
}
